// GCN_16518444221106
// MI455X (gfx1250) — matrix operand order repaired and checked by reading, not compiled, not run on hardware
//
#include <hip/hip_runtime.h>
#include <hip/hip_bf16.h>

#define N_NODES 50000
#define N_PAD   50016      // multiple of 32
#define F_IN    500
#define KP1     512        // F_IN padded to multiple of 32
#define H_DIM   256
#define C_OUTN  40
#define C_PADN  48         // 40 padded to multiple of 16
#define K_EIG   128
#define BN_EPS  1e-5f

typedef __attribute__((ext_vector_type(16))) __bf16          v16bf;
typedef __attribute__((ext_vector_type(8)))  float           v8f;
typedef __attribute__((ext_vector_type(16))) unsigned short  v16u;

__device__ __forceinline__ unsigned short f2bf(float f) {
  unsigned u = __float_as_uint(f);
  u += 0x7FFFu + ((u >> 16) & 1u);        // round-to-nearest-even
  return (unsigned short)(u >> 16);
}
__device__ __forceinline__ float bf2f(unsigned short h) {
  return __uint_as_float(((unsigned)h) << 16);
}

// ---------------------------------------------------------------------------
// Convert fp32 [Nrows x Ks] -> bf16 [N_PAD x Kp], zero padding rows/cols.
// ---------------------------------------------------------------------------
__global__ void cvt_pad_bf16(const float* __restrict__ src, unsigned short* __restrict__ dst,
                             int Ks, int Kp, int Nrows) {
  long idx = (long)blockIdx.x * 256 + threadIdx.x;
  long total = (long)N_PAD * Kp;
  if (idx >= total) return;
  int i = (int)(idx / Kp);
  int k = (int)(idx - (long)i * Kp);
  float v = (i < Nrows && k < Ks) ? src[(long)i * Ks + k] : 0.f;
  dst[idx] = f2bf(v);
}

// W [Ks x Nsrc] fp32 -> Wt [Ncp x Kp] bf16 (transposed, zero-padded)
__global__ void cvt_w_t(const float* __restrict__ W, unsigned short* __restrict__ Wt,
                        int Ks, int Nsrc, int Kp, int Ncp) {
  int idx = blockIdx.x * 256 + threadIdx.x;
  if (idx >= Ncp * Kp) return;
  int n = idx / Kp;
  int k = idx - n * Kp;
  float v = (k < Ks && n < Nsrc) ? W[(long)k * Nsrc + n] : 0.f;
  Wt[idx] = f2bf(v);
}

__global__ void zero_f32(float* __restrict__ p, long n) {
  long i = (long)blockIdx.x * 256 + threadIdx.x;
  if (i < n) p[i] = 0.f;
}

// ---------------------------------------------------------------------------
// WMMA GEMM, register-blocked: one wave computes a 16 x (16*NB) strip.
// C[M x Ncp] = A[~ x Kd](bf16, ld=lda) * Bt[Ncp x Kd](bf16, ld=ldb)^T
// Fragments per CDNA5 ISA 7.12.2 layouts:
//   A 16x32 bf16: lanes 0-15 row M=l, elems 0..7 -> K=k0+8h.., elems 8..15 -> K=k0+16+8h..
//   B 32x16 bf16: lane col N=l, elems 0..7 -> K=k0+8h.., elems 8..15 -> K=k0+16+8h..   (Bt row-contiguous)
//   C 16x16 f32:  lane half h, col=l, VGPR r -> row 8h+r
// OUT_BF16: store bf16 tile.  ACC_BIAS: C += existing + bias[col<nbias].
// ---------------------------------------------------------------------------
template <int NB, bool OUT_BF16, bool ACC_BIAS>
__global__ void wmma_gemm(const unsigned short* __restrict__ A, int lda,
                          const unsigned short* __restrict__ Bt, int ldb,
                          void* __restrict__ C, int ldc, int Kd,
                          const float* __restrict__ bias, int nbias) {
  int lane = threadIdx.x;              // 32 threads = 1 wave, EXEC all ones
  int half = lane >> 4;
  int l    = lane & 15;
  long m0 = (long)blockIdx.x * 16;
  long n0 = (long)blockIdx.y * (16 * NB);

  const unsigned short* arow = A + (m0 + l) * (long)lda;
  const unsigned short* brow[NB];
#pragma unroll
  for (int j = 0; j < NB; ++j) brow[j] = Bt + (n0 + 16 * j + l) * (long)ldb;

  v8f acc[NB];
#pragma unroll
  for (int j = 0; j < NB; ++j) acc[j] = (v8f){0.f, 0.f, 0.f, 0.f, 0.f, 0.f, 0.f, 0.f};

  for (int k0 = 0; k0 < Kd; k0 += 32) {
    if (k0 + 32 < Kd) __builtin_prefetch(arow + k0 + 32, 0, 1);  // global_prefetch_b8
    union { uint4 q[2]; v16u v; } fa;
    fa.q[0] = *(const uint4*)(arow + k0 + 8 * half);        // K = k0+8h .. +7
    fa.q[1] = *(const uint4*)(arow + k0 + 16 + 8 * half);   // K = k0+16+8h ..
    v16bf av = __builtin_bit_cast(v16bf, fa.v);
#pragma unroll
    for (int j = 0; j < NB; ++j) {
      union { uint4 q[2]; v16u v; } fb;
      fb.q[0] = *(const uint4*)(brow[j] + k0 + 8 * half);        // K = k0+8h .. +7
      fb.q[1] = *(const uint4*)(brow[j] + k0 + 16 + 8 * half);   // K = k0+16+8h ..
      acc[j] = __builtin_amdgcn_wmma_f32_16x16x32_bf16(
          false, av, false, __builtin_bit_cast(v16bf, fb.v),
          (short)0, acc[j], false, false);
    }
  }

#pragma unroll
  for (int j = 0; j < NB; ++j) {
    long col = n0 + 16 * j + l;
    if constexpr (OUT_BF16) {
      unsigned short* out = (unsigned short*)C;
#pragma unroll
      for (int r = 0; r < 8; ++r)
        out[(m0 + 8 * half + r) * (long)ldc + col] = f2bf(acc[j][r]);
    } else {
      float* out = (float*)C;
      float bv = 0.f;
      if constexpr (ACC_BIAS) bv = (col < nbias) ? bias[col] : 0.f;
#pragma unroll
      for (int r = 0; r < 8; ++r) {
        long o = (m0 + 8 * half + r) * (long)ldc + col;
        float v = acc[j][r] + bv;
        if constexpr (ACC_BIAS) v += out[o];
        out[o] = v;
      }
    }
  }
}

// ---------------------------------------------------------------------------
// tilda[k][n] = sum_i ev[i][k] * support[i][n]   (reduction dim = N nodes)
// LDS-tiled outer product, fp32 atomics for split-N partials.
// ---------------------------------------------------------------------------
__global__ void tilda_partial(const float* __restrict__ ev,
                              const unsigned short* __restrict__ supb,
                              float* __restrict__ tilda, int Ncp, int rows_per_blk) {
  __shared__ float sev[K_EIG];
  __shared__ float ssup[64];
  int t    = threadIdx.x;
  int n0   = blockIdx.x * 64;
  int nloc = t & 63;
  int kgrp = t >> 6;                       // 0..3
  int r0 = blockIdx.y * rows_per_blk;
  int r1 = r0 + rows_per_blk;
  if (r1 > N_NODES) r1 = N_NODES;

  float acc[32];
#pragma unroll
  for (int j = 0; j < 32; ++j) acc[j] = 0.f;

  int  ncol = n0 + nloc;
  bool nok  = ncol < Ncp;

  for (int i = r0; i < r1; ++i) {
    if (t < 128) {
      sev[t] = ev[(long)i * K_EIG + t];
    } else if (t < 192) {
      int nn = n0 + (t - 128);
      ssup[t - 128] = (nn < Ncp) ? bf2f(supb[(long)i * Ncp + nn]) : 0.f;
    }
    __syncthreads();
    float sv = ssup[nloc];
#pragma unroll
    for (int j = 0; j < 32; ++j) acc[j] += sev[kgrp + 4 * j] * sv;
    __syncthreads();
  }
  if (nok) {
#pragma unroll
    for (int j = 0; j < 32; ++j)
      atomicAdd(&tilda[(kgrp + 4 * j) * Ncp + ncol], acc[j]);
  }
}

// Mt[n][k] = bf16(delta[k] * tilda[k][n])   (pre-transposed B for WMMA GEMM2)
__global__ void make_mt(const float* __restrict__ tilda, const float* __restrict__ delta,
                        unsigned short* __restrict__ Mt, int Ncp) {
  int idx = blockIdx.x * 256 + threadIdx.x;
  if (idx >= Ncp * K_EIG) return;
  int n = idx >> 7;
  int k = idx & 127;
  Mt[idx] = f2bf(delta[k] * tilda[k * Ncp + n]);
}

// COO SpMM (Ncp==256): wave-per-edge, lane covers 8 contiguous bf16 cols (b128 load).
__global__ void spmm_vec256(const int* __restrict__ rows, const int* __restrict__ cols,
                            const float* __restrict__ vals,
                            const unsigned short* __restrict__ supb,
                            float* __restrict__ agg, int E) {
  int wave = threadIdx.x >> 5;
  int lane = threadIdx.x & 31;
  long e = (long)blockIdx.x * 8 + wave;
  if (e >= E) return;
  int r = rows[e];
  int s = cols[e];
  float v = vals[e];
  int c0 = lane * 8;
  union { uint4 q; unsigned short u[8]; } d;
  d.q = *(const uint4*)(supb + (long)s * 256 + c0);
  float* dst = agg + (long)r * 256 + c0;
#pragma unroll
  for (int j = 0; j < 8; ++j) atomicAdd(dst + j, v * bf2f(d.u[j]));
}

// COO SpMM scalar fallback (layer 3, Ncp=48): block per edge.
__global__ void spmm(const int* __restrict__ rows, const int* __restrict__ cols,
                     const float* __restrict__ vals,
                     const unsigned short* __restrict__ supb,
                     float* __restrict__ agg, int Ncp) {
  int e = blockIdx.x;
  int c = threadIdx.x;
  if (c >= Ncp) return;
  int r = rows[e];
  int s = cols[e];
  float v = vals[e];
  atomicAdd(&agg[(long)r * Ncp + c], v * bf2f(supb[(long)s * Ncp + c]));
}

// BN partial sums: thread t owns channel t; coalesced row streaming + atomics.
__global__ void bn_partial(const float* __restrict__ h, float* __restrict__ gsum,
                           float* __restrict__ gsq, int rows_per_blk) {
  int t = threadIdx.x;                 // 256 == H_DIM
  int r0 = blockIdx.x * rows_per_blk;
  int r1 = r0 + rows_per_blk;
  if (r1 > N_NODES) r1 = N_NODES;
  float sum = 0.f, sq = 0.f;
  for (int i = r0; i < r1; ++i) {
    float v = h[(long)i * H_DIM + t];
    sum += v;
    sq  += v * v;
  }
  atomicAdd(&gsum[t], sum);
  atomicAdd(&gsq[t], sq);
}

__global__ void bn_finalize(const float* __restrict__ gsum, const float* __restrict__ gsq,
                            const float* __restrict__ gamma, const float* __restrict__ beta,
                            float* __restrict__ scale, float* __restrict__ shift) {
  int c = threadIdx.x;                 // 256
  float mean = gsum[c] / (float)N_NODES;
  float var  = gsq[c] / (float)N_NODES - mean * mean;
  float sc   = rsqrtf(var + BN_EPS) * gamma[c];
  scale[c] = sc;
  shift[c] = beta[c] - mean * sc;
}

// h -> relu(h*scale+shift) -> bf16 A-matrix for next layer (pad rows zeroed)
__global__ void bn_apply_relu_bf16(const float* __restrict__ h,
                                   const float* __restrict__ scale,
                                   const float* __restrict__ shift,
                                   unsigned short* __restrict__ actb) {
  long idx = (long)blockIdx.x * 256 + threadIdx.x;
  if (idx >= (long)N_PAD * H_DIM) return;
  int i = (int)(idx >> 8);
  int c = (int)(idx & 255);
  float v = 0.f;
  if (i < N_NODES) {
    v = h[idx] * scale[c] + shift[c];
    v = fmaxf(v, 0.f);
  }
  actb[idx] = f2bf(v);
}

// Row-wise log_softmax over 40 valid columns (stride C_PADN); block = 64.
__global__ void log_softmax40(const float* __restrict__ h, float* __restrict__ out) {
  __shared__ float s[64];
  int i = blockIdx.x;
  int t = threadIdx.x;
  float v = (t < C_OUTN) ? h[(long)i * C_PADN + t] : -3.4e38f;
  s[t] = v;
  __syncthreads();
  for (int w = 32; w > 0; w >>= 1) { if (t < w) s[t] = fmaxf(s[t], s[t + w]); __syncthreads(); }
  float mx = s[0];
  __syncthreads();
  float ex = (t < C_OUTN) ? __expf(v - mx) : 0.f;
  s[t] = ex;
  __syncthreads();
  for (int w = 32; w > 0; w >>= 1) { if (t < w) s[t] += s[t + w]; __syncthreads(); }
  float lse = logf(s[0]);
  if (t < C_OUTN) out[(long)i * C_OUTN + t] = v - mx - lse;
}

// ---------------------------------------------------------------------------
struct LayerBufs {
  const int* rows; const int* cols; const float* vals;
  const float* ev; const float* delta;
  unsigned short* supb; unsigned short* evb; unsigned short* Wt;
  float* tild; unsigned short* Mt; int E;
};

template <int NB>
static void run_layer(const LayerBufs& B, const unsigned short* Abf, int Kp,
                      const float* W, int Ks, int Nsrc, int Ncp,
                      const float* bias, int nbias, float* outh, hipStream_t stream) {
  // weights transposed -> bf16
  int wtot = Ncp * Kp;
  cvt_w_t<<<(wtot + 255) / 256, 256, 0, stream>>>(W, B.Wt, Ks, Nsrc, Kp, Ncp);
  // GEMM1 (WMMA bf16): supb = A @ W   (pad rows come out zero: A pad rows are zero)
  dim3 g1g(N_PAD / 16, Ncp / (16 * NB));
  wmma_gemm<NB, true, false><<<g1g, 32, 0, stream>>>(Abf, Kp, B.Wt, Kp, B.supb, Ncp, Kp,
                                                     nullptr, 0);
  // tilda = ev^T @ support  (split-N partials + atomics)
  zero_f32<<<(K_EIG * Ncp + 255) / 256, 256, 0, stream>>>(B.tild, (long)K_EIG * Ncp);
  dim3 tg((Ncp + 63) / 64, 100);
  tilda_partial<<<tg, 256, 0, stream>>>(B.ev, B.supb, B.tild, Ncp, 500);
  // Mt = (delta * tilda)^T in bf16  (must precede GEMM2)
  make_mt<<<(Ncp * K_EIG + 255) / 256, 256, 0, stream>>>(B.tild, B.delta, B.Mt, Ncp);
  // agg = SpMM
  long atot = (long)N_NODES * Ncp;
  zero_f32<<<(unsigned)((atot + 255) / 256), 256, 0, stream>>>(outh, atot);
  if (Ncp == 256) {
    spmm_vec256<<<(B.E + 7) / 8, 256, 0, stream>>>(B.rows, B.cols, B.vals, B.supb, outh, B.E);
  } else {
    spmm<<<B.E, 64, 0, stream>>>(B.rows, B.cols, B.vals, B.supb, outh, Ncp);
  }
  // GEMM2 (WMMA bf16): outh += evb @ Mt^T + bias
  dim3 g2g(N_NODES / 16, Ncp / (16 * NB));
  wmma_gemm<NB, false, true><<<g2g, 32, 0, stream>>>(B.evb, K_EIG, B.Mt, K_EIG, outh, Ncp,
                                                     K_EIG, bias, nbias);
}

extern "C" void kernel_launch(void* const* d_in, const int* in_sizes, int n_in,
                              void* d_out, int out_size, void* d_ws, size_t ws_size,
                              hipStream_t stream) {
  const float* x    = (const float*)d_in[0];
  const int*   rows = (const int*)d_in[1];
  const int*   cols = (const int*)d_in[2];
  const float* vals = (const float*)d_in[3];
  const float* ev   = (const float*)d_in[4];
  const float* w1   = (const float*)d_in[5];
  const float* b1   = (const float*)d_in[6];
  const float* g1   = (const float*)d_in[7];
  const float* be1  = (const float*)d_in[8];
  const float* w2   = (const float*)d_in[9];
  const float* b2   = (const float*)d_in[10];
  const float* g2   = (const float*)d_in[11];
  const float* be2  = (const float*)d_in[12];
  const float* w3   = (const float*)d_in[13];
  const float* b3   = (const float*)d_in[14];
  const float* delta = (const float*)d_in[15];
  const int E = in_sizes[1];

  // workspace carve-out (all 256B aligned)
  char*  ws  = (char*)d_ws;
  size_t off = 0;
  auto carve = [&](size_t bytes) -> char* {
    char* p = ws + off;
    off = (off + bytes + 255) & ~(size_t)255;
    return p;
  };
  unsigned short* actb = (unsigned short*)carve((size_t)N_PAD * KP1 * 2);   // 51.2 MB
  unsigned short* evb  = (unsigned short*)carve((size_t)N_PAD * K_EIG * 2); // 12.8 MB
  unsigned short* supb = (unsigned short*)carve((size_t)N_PAD * H_DIM * 2); // 25.6 MB
  float*          agg  = (float*)carve((size_t)N_NODES * H_DIM * 4);        // 51.2 MB
  unsigned short* Wt   = (unsigned short*)carve((size_t)H_DIM * KP1 * 2);
  float*          tild = (float*)carve((size_t)K_EIG * H_DIM * 4);
  unsigned short* Mt   = (unsigned short*)carve((size_t)H_DIM * K_EIG * 2);
  float*          bnsc = (float*)carve(256 * 4);
  float*          bnsh = (float*)carve(256 * 4);
  float*          bnsum = (float*)carve(256 * 4);
  float*          bnsq  = (float*)carve(256 * 4);
  (void)ws_size; (void)n_in; (void)out_size;

  LayerBufs B{rows, cols, vals, ev, delta, supb, evb, Wt, tild, Mt, E};

  // input conversions
  {
    long t1 = (long)N_PAD * KP1;
    cvt_pad_bf16<<<(unsigned)((t1 + 255) / 256), 256, 0, stream>>>(x, actb, F_IN, KP1, N_NODES);
    long t2 = (long)N_PAD * K_EIG;
    cvt_pad_bf16<<<(unsigned)((t2 + 255) / 256), 256, 0, stream>>>(ev, evb, K_EIG, K_EIG, N_NODES);
  }

  auto do_bn = [&](const float* gamma, const float* beta) {
    zero_f32<<<2, 256, 0, stream>>>(bnsum, 256);
    zero_f32<<<2, 256, 0, stream>>>(bnsq, 256);
    bn_partial<<<100, 256, 0, stream>>>(agg, bnsum, bnsq, 500);
    bn_finalize<<<1, 256, 0, stream>>>(bnsum, bnsq, gamma, beta, bnsc, bnsh);
    long t = (long)N_PAD * H_DIM;
    bn_apply_relu_bf16<<<(unsigned)((t + 255) / 256), 256, 0, stream>>>(agg, bnsc, bnsh, actb);
  };

  // layer 1: [N x 500] -> [N x 256], BN+ReLU
  run_layer<4>(B, actb, KP1, w1, F_IN, H_DIM, H_DIM, b1, H_DIM, agg, stream);
  do_bn(g1, be1);

  // layer 2: [N x 256] -> [N x 256], BN+ReLU
  run_layer<4>(B, actb, H_DIM, w2, H_DIM, H_DIM, H_DIM, b2, H_DIM, agg, stream);
  do_bn(g2, be2);

  // layer 3: [N x 256] -> [N x 48(pad of 40)], then log_softmax over 40
  run_layer<3>(B, actb, H_DIM, w3, H_DIM, C_OUTN, C_PADN, b3, C_OUTN, agg, stream);
  log_softmax40<<<N_NODES, 64, 0, stream>>>(agg, (float*)d_out);
}
